// SparseAttention_34488587387650
// MI455X (gfx1250) — matrix operand order repaired and checked by reading, not compiled, not run on hardware
//
#include <hip/hip_runtime.h>

typedef __attribute__((ext_vector_type(16))) __bf16 v16bf;
typedef __attribute__((ext_vector_type(8)))  __bf16 v8bf;
typedef __attribute__((ext_vector_type(8)))  float  v8f;
typedef __attribute__((ext_vector_type(4)))  int    v4i;

#define HIDDEN   1024
#define HEADS    16
#define HEAD_DIM 64
#define BATCH    2
#define SEQ      2048
#define M_TOTAL  (BATCH*SEQ)   // 4096

#define WMMA_BF16(A,B,C) \
  __builtin_amdgcn_wmma_f32_16x16x32_bf16(false,(A),false,(B),(short)0,(C),false,false)

static __device__ __forceinline__ __bf16 f2bf(float f) {
  unsigned u = __builtin_bit_cast(unsigned, f);
  u += 0x7FFFu + ((u >> 16) & 1u);               // round-to-nearest-even
  unsigned short h = (unsigned short)(u >> 16);
  return __builtin_bit_cast(__bf16, h);
}

union V16U { v16bf v; v8bf h[2]; };

// ---- async global->LDS copy (16B), ASYNCcnt-tracked; sync fallback ----
static __device__ __forceinline__ void async_copy16(const __bf16* g, __bf16* l) {
#if __has_builtin(__builtin_amdgcn_global_load_async_to_lds_b128)
  __builtin_amdgcn_global_load_async_to_lds_b128(
      (__attribute__((address_space(1))) v4i*)(v4i*)g,
      (__attribute__((address_space(3))) v4i*)(v4i*)l, 0, 0);
#else
  *(v8bf*)l = *(const v8bf*)g;
#endif
}

static __device__ __forceinline__ void async_wait_all() {
#if __has_builtin(__builtin_amdgcn_s_wait_asynccnt)
  __builtin_amdgcn_s_wait_asynccnt(0);
#elif __has_builtin(__builtin_amdgcn_global_load_async_to_lds_b128)
  asm volatile("s_wait_asynccnt 0x0" ::: "memory");
#endif
}

// ---------------- fp32 -> bf16 elementwise ----------------
__global__ void cvt_x_kernel(const float* __restrict__ in, __bf16* __restrict__ out, int n) {
  int i = blockIdx.x * blockDim.x + threadIdx.x;
  if (i < n) out[i] = f2bf(in[i]);
}

// ---------------- W[K][N] fp32 -> Wt[N][K] bf16 (LDS-tiled transpose) ----------------
__global__ void cvt_wT_kernel(const float* __restrict__ W, __bf16* __restrict__ Wt) {
  __shared__ float tile[32][33];
  int n0 = blockIdx.x * 32, k0 = blockIdx.y * 32;
  int tx = threadIdx.x, ty = threadIdx.y;           // block (32,8)
  for (int i = ty; i < 32; i += 8)
    tile[i][tx] = W[(size_t)(k0 + i) * HIDDEN + n0 + tx];
  __syncthreads();
  for (int i = ty; i < 32; i += 8)
    Wt[(size_t)(n0 + i) * HIDDEN + k0 + tx] = f2bf(tile[tx][i]);
}

// ---- stage Wt[n0 .. n0+63][k0 .. k0+31] into LDS (64 rows x 32 el) ----
static __device__ __forceinline__ void gemm_stage(const __bf16* __restrict__ Wt,
                                                  __bf16* Bs, int n0, int k0, int tid) {
  #pragma unroll
  for (int c = 0; c < 2; ++c) {
    int chunk = tid * 2 + c;            // 0..255 (16B chunks)
    int row = chunk >> 2;               // 0..63
    int off = (chunk & 3) * 8;          // 0,8,16,24 elements
    async_copy16(Wt + (size_t)(n0 + row) * HIDDEN + k0 + off, Bs + row * 32 + off);
  }
}

// ---------------- bf16 WMMA GEMM: Y[M][1024] = X[M][1024] @ Wt^T + bias ----------------
// Block: 4 waves, each wave 32(M) x 64(N). B panel staged in LDS, double-buffered async.
// mode 0: Q -> bf16 [bh][s][d] *0.125 | 1: K -> [bh][s][d] | 2: V -> [bh][d][s] | 3: fp32 out
__global__ __launch_bounds__(128)
void gemm16_kernel(const __bf16* __restrict__ X, const __bf16* __restrict__ Wt,
                   const float* __restrict__ bias, void* __restrict__ outp, int mode) {
  __shared__ __align__(128) __bf16 Bs[2][64 * 32];
  int tid = threadIdx.x;
  int lane = tid & 31, wave = tid >> 5;
  int hl = lane >> 4, lm = lane & 15;
  int m0 = blockIdx.y * 128 + wave * 32;
  int n0 = blockIdx.x * 64;

  v8f acc[2][4] = {};
  const __bf16* xrow0 = X + (size_t)(m0 + lm) * HIDDEN + 16 * hl;
  const __bf16* xrow1 = xrow0 + 16 * HIDDEN;

  gemm_stage(Wt, Bs[0], n0, 0, tid);
  for (int it = 0; it < HIDDEN / 32; ++it) {
    int k0 = it * 32;
    const __bf16* bs = Bs[it & 1];
    async_wait_all();
    __syncthreads();                                   // panel ready; prev reads done
    if (it + 1 < HIDDEN / 32) gemm_stage(Wt, Bs[(it + 1) & 1], n0, k0 + 32, tid);

    V16U a0, a1;                                       // A: 2x (16x32) tiles
    a0.h[0] = *(const v8bf*)(xrow0 + k0);
    a0.h[1] = *(const v8bf*)(xrow0 + k0 + 8);
    a1.h[0] = *(const v8bf*)(xrow1 + k0);
    a1.h[1] = *(const v8bf*)(xrow1 + k0 + 8);
    #pragma unroll
    for (int j = 0; j < 4; ++j) {
      v16bf b = *(const v16bf*)(bs + (16 * j + lm) * 32 + 16 * hl);
      acc[0][j] = WMMA_BF16(a0.v, b, acc[0][j]);
      acc[1][j] = WMMA_BF16(a1.v, b, acc[1][j]);
    }
    __syncthreads();                                   // protect buffer being refilled
  }

  #pragma unroll
  for (int i = 0; i < 2; ++i) {
    #pragma unroll
    for (int j = 0; j < 4; ++j) {
      int n = n0 + 16 * j + lm;
      float bn = bias[n];
      #pragma unroll
      for (int r = 0; r < 8; ++r) {
        int m = m0 + 16 * i + r + 8 * hl;
        float val = acc[i][j][r] + bn;
        if (mode == 3) {
          ((float*)outp)[(size_t)m * HIDDEN + n] = val;
        } else {
          int b = m >> 11, s = m & 2047;
          int hh = n >> 6, d = n & 63;
          __bf16* o = (__bf16*)outp;
          if (mode == 0)
            o[((size_t)(b * HEADS + hh) * SEQ + s) * HEAD_DIM + d] = f2bf(val * 0.125f);
          else if (mode == 1)
            o[((size_t)(b * HEADS + hh) * SEQ + s) * HEAD_DIM + d] = f2bf(val);
          else
            o[((size_t)(b * HEADS + hh) * HEAD_DIM + d) * SEQ + s] = f2bf(val);
        }
      }
    }
  }
}

// ---- stage K tile (32 x 64, row-major) and V^T tile (64 x 32) into LDS ----
static __device__ __forceinline__ void fa_stage(const __bf16* __restrict__ Kh,
                                                const __bf16* __restrict__ Vh,
                                                __bf16* Ks, __bf16* Vs, int t0, int tid) {
  #pragma unroll
  for (int c = 0; c < 2; ++c) {
    int chunk = tid * 2 + c;            // 0..255
    int kr = chunk >> 3, ko = (chunk & 7) * 8;     // K: 8 chunks per 64-el row
    async_copy16(Kh + (size_t)(t0 + kr) * HEAD_DIM + ko, Ks + kr * 64 + ko);
    int vr = chunk >> 2, vo = (chunk & 3) * 8;     // V: 4 chunks per 32-el row
    async_copy16(Vh + (size_t)vr * SEQ + t0 + vo, Vs + vr * 32 + vo);
  }
}

// ---------------- flash attention: block = 4 waves x 16 queries, shared K/V LDS tiles ----------------
__global__ __launch_bounds__(128)
void fa_kernel(const __bf16* __restrict__ Q, const __bf16* __restrict__ K,
               const __bf16* __restrict__ Vt, __bf16* __restrict__ AO) {
  __shared__ __align__(128) __bf16 Ks[2][32 * 64];
  __shared__ __align__(128) __bf16 Vs[2][64 * 32];
  int tid = threadIdx.x;
  int lane = tid & 31, wave = tid >> 5;
  int hl = lane >> 4, lm = lane & 15;
  int bh = blockIdx.y;                      // b*16+h
  int sq0 = blockIdx.x * 64 + wave * 16;

  const __bf16* Qh = Q  + (size_t)bh * SEQ * HEAD_DIM;
  const __bf16* Kh = K  + (size_t)bh * SEQ * HEAD_DIM;
  const __bf16* Vh = Vt + (size_t)bh * HEAD_DIM * SEQ;

  // Q as B-matrix of scoresT: lane holds query s=lm, contiguous d
  const __bf16* qrow = Qh + (size_t)(sq0 + lm) * HEAD_DIM;
  v16bf bq0 = *(const v16bf*)(qrow + 16 * hl);
  v16bf bq1 = *(const v16bf*)(qrow + 32 + 16 * hl);

  float m_run = -1e30f, l_run = 0.f;
  v8f O0 = {}, O1 = {}, O2 = {}, O3 = {};

  fa_stage(Kh, Vh, Ks[0], Vs[0], 0, tid);
  for (int it = 0; it < SEQ / 32; ++it) {
    int buf = it & 1;
    async_wait_all();
    __syncthreads();
    if (it + 1 < SEQ / 32) fa_stage(Kh, Vh, Ks[buf ^ 1], Vs[buf ^ 1], (it + 1) * 32, tid);

    // K rows as A-matrix fragments from LDS
    const __bf16* kb0 = Ks[buf] + lm * 64 + 16 * hl;
    const __bf16* kb1 = Ks[buf] + (16 + lm) * 64 + 16 * hl;
    V16U a00, a01, a10, a11;
    a00.h[0] = *(const v8bf*)(kb0);      a00.h[1] = *(const v8bf*)(kb0 + 8);
    a01.h[0] = *(const v8bf*)(kb0 + 32); a01.h[1] = *(const v8bf*)(kb0 + 40);
    a10.h[0] = *(const v8bf*)(kb1);      a10.h[1] = *(const v8bf*)(kb1 + 8);
    a11.h[0] = *(const v8bf*)(kb1 + 32); a11.h[1] = *(const v8bf*)(kb1 + 40);

    v8f st0 = {}, st1 = {};                             // scoresT: [t][s]
    st0 = WMMA_BF16(a00.v, bq0, st0); st0 = WMMA_BF16(a01.v, bq1, st0);
    st1 = WMMA_BF16(a10.v, bq0, st1); st1 = WMMA_BF16(a11.v, bq1, st1);

    // online softmax: lane holds fixed s=lm, 16 t-values; other t-half in lane^16
    float mp = -1e30f;
    #pragma unroll
    for (int r = 0; r < 8; ++r) { mp = fmaxf(mp, st0[r]); mp = fmaxf(mp, st1[r]); }
    mp = fmaxf(mp, __shfl_xor(mp, 16, 32));
    float m_new = fmaxf(m_run, mp);
    float alpha = __expf(m_run - m_new);

    float p0[8], p1[8], psum = 0.f;
    #pragma unroll
    for (int r = 0; r < 8; ++r) {
      p0[r] = __expf(st0[r] - m_new);
      p1[r] = __expf(st1[r] - m_new);
      psum += p0[r] + p1[r];
    }
    psum += __shfl_xor(psum, 16, 32);
    l_run = l_run * alpha + psum;
    m_run = m_new;

    // scoresT C-fragment -> P A-fragment: element-wise copy, no cross-lane transpose
    V16U ap;
    #pragma unroll
    for (int r = 0; r < 8; ++r) { ap.v[r] = f2bf(p0[r]); ap.v[r + 8] = f2bf(p1[r]); }

    // rescale O: alpha for row s=r+8*hl lives in lane (r+8*hl)
    #pragma unroll
    for (int r = 0; r < 8; ++r) {
      float ar = __shfl(alpha, r + 8 * hl, 32);
      O0[r] *= ar; O1[r] *= ar; O2[r] *= ar; O3[r] *= ar;
    }

    // V^T as B-matrix fragments from LDS: lane holds fixed d, t in table order (8*hl, 16+8*hl)
    V16U vb0, vb1, vb2, vb3;
    vb0.h[0] = *(const v8bf*)(Vs[buf] + (     lm) * 32 + 8 * hl); vb0.h[1] = *(const v8bf*)(Vs[buf] + (     lm) * 32 + 16 + 8 * hl);
    vb1.h[0] = *(const v8bf*)(Vs[buf] + (16 + lm) * 32 + 8 * hl); vb1.h[1] = *(const v8bf*)(Vs[buf] + (16 + lm) * 32 + 16 + 8 * hl);
    vb2.h[0] = *(const v8bf*)(Vs[buf] + (32 + lm) * 32 + 8 * hl); vb2.h[1] = *(const v8bf*)(Vs[buf] + (32 + lm) * 32 + 16 + 8 * hl);
    vb3.h[0] = *(const v8bf*)(Vs[buf] + (48 + lm) * 32 + 8 * hl); vb3.h[1] = *(const v8bf*)(Vs[buf] + (48 + lm) * 32 + 16 + 8 * hl);
    v16bf b0 = vb0.v, b1 = vb1.v, b2 = vb2.v, b3 = vb3.v;
    O0 = WMMA_BF16(ap.v, b0, O0); O1 = WMMA_BF16(ap.v, b1, O1);
    O2 = WMMA_BF16(ap.v, b2, O2); O3 = WMMA_BF16(ap.v, b3, O3);

    __syncthreads();                                    // protect buffer being refilled
  }

  int b = bh >> 4, h = bh & 15;
  #pragma unroll
  for (int r = 0; r < 8; ++r) {
    float lr  = __shfl(l_run, r + 8 * hl, 32);
    float inv = 1.f / lr;
    int sg = sq0 + r + 8 * hl;
    __bf16* orow = AO + (size_t)(b * SEQ + sg) * HIDDEN + h * HEAD_DIM + lm;
    orow[0]  = f2bf(O0[r] * inv);
    orow[16] = f2bf(O1[r] * inv);
    orow[32] = f2bf(O2[r] * inv);
    orow[48] = f2bf(O3[r] * inv);
  }
}

extern "C" void kernel_launch(void* const* d_in, const int* in_sizes, int n_in,
                              void* d_out, int out_size, void* d_ws, size_t ws_size,
                              hipStream_t stream) {
  const float* x  = (const float*)d_in[0];
  const float* Wq = (const float*)d_in[1];
  const float* bq = (const float*)d_in[2];
  const float* Wk = (const float*)d_in[3];
  const float* bk = (const float*)d_in[4];
  const float* Wv = (const float*)d_in[5];
  const float* bv = (const float*)d_in[6];
  const float* Wo = (const float*)d_in[7];
  const float* bo = (const float*)d_in[8];
  float* out = (float*)d_out;

  __bf16* ws = (__bf16*)d_ws;
  size_t o = 0;
  __bf16* xb  = ws + o; o += (size_t)M_TOTAL * HIDDEN;
  __bf16* Wqt = ws + o; o += (size_t)HIDDEN * HIDDEN;
  __bf16* Wkt = ws + o; o += (size_t)HIDDEN * HIDDEN;
  __bf16* Wvt = ws + o; o += (size_t)HIDDEN * HIDDEN;
  __bf16* Wot = ws + o; o += (size_t)HIDDEN * HIDDEN;
  __bf16* Qb  = ws + o; o += (size_t)M_TOTAL * HIDDEN;
  __bf16* Kb  = ws + o; o += (size_t)M_TOTAL * HIDDEN;
  __bf16* Vtb = ws + o; o += (size_t)M_TOTAL * HIDDEN;
  __bf16* AO  = ws + o; o += (size_t)M_TOTAL * HIDDEN;   // 48 MiB total

  int nx = M_TOTAL * HIDDEN;
  cvt_x_kernel<<<(nx + 255) / 256, 256, 0, stream>>>(x, xb, nx);
  dim3 tg(32, 32), tb(32, 8);
  cvt_wT_kernel<<<tg, tb, 0, stream>>>(Wq, Wqt);
  cvt_wT_kernel<<<tg, tb, 0, stream>>>(Wk, Wkt);
  cvt_wT_kernel<<<tg, tb, 0, stream>>>(Wv, Wvt);
  cvt_wT_kernel<<<tg, tb, 0, stream>>>(Wo, Wot);

  dim3 gg(HIDDEN / 64, M_TOTAL / 128);  // (16, 32)
  gemm16_kernel<<<gg, 128, 0, stream>>>(xb, Wqt, bq, Qb, 0);
  gemm16_kernel<<<gg, 128, 0, stream>>>(xb, Wkt, bk, Kb, 1);
  gemm16_kernel<<<gg, 128, 0, stream>>>(xb, Wvt, bv, Vtb, 2);

  fa_kernel<<<dim3(SEQ / 64, BATCH * HEADS), 128, 0, stream>>>(Qb, Kb, Vtb, AO);

  gemm16_kernel<<<gg, 128, 0, stream>>>(AO, Wot, bo, out, 3);
}
